// TemporalGNN_11836929868099
// MI455X (gfx1250) — matrix operand order repaired and checked by reading, not compiled, not run on hardware
//
#include <hip/hip_runtime.h>
#include <math.h>

constexpr int T_ = 8, N_ = 1024, IN_ = 64, HID_ = 128, OUT_ = 64, E_ = 16384;

typedef __attribute__((ext_vector_type(16))) _Float16 v16h;
typedef __attribute__((ext_vector_type(8)))  _Float16 v8h;
typedef __attribute__((ext_vector_type(8)))  float    v8f;

// ---------------- degree ----------------
__global__ void k_init_deg(float* deg) {
    int i = blockIdx.x * 256 + threadIdx.x;
    if (i < T_ * N_) deg[i] = 1.0f;
}

__global__ void k_count_deg(const int* __restrict__ ei, float* __restrict__ deg) {
    int g = blockIdx.x * 256 + threadIdx.x;
    if (g >= T_ * E_) return;
    int t = g / E_, e = g % E_;
    int dst = ei[t * 2 * E_ + E_ + e];
    atomicAdd(&deg[t * N_ + dst], 1.0f);
}

__global__ void k_fin_deg(float* __restrict__ deg, float* __restrict__ dinv) {
    int i = blockIdx.x * 256 + threadIdx.x;
    if (i >= T_ * N_) return;
    float d = deg[i];
    deg[i]  = 1.0f / d;        // buffer becomes 1/deg
    dinv[i] = rsqrtf(d);       // deg^-1/2
}

// ---------------- f16 packing ----------------
__global__ void k_pack_f16(const float* __restrict__ src, _Float16* __restrict__ dst, int n) {
    int i = blockIdx.x * 256 + threadIdx.x;
    if (i < n) dst[i] = (_Float16)src[i];
}

// B (K x NC, row-major f32) -> BT (NC x K, row-major f16): lane fragment = contiguous run
__global__ void k_packBT(const float* __restrict__ B, _Float16* __restrict__ BT,
                         int K, int NC) {
    int i = blockIdx.x * 256 + threadIdx.x;
    if (i >= K * NC) return;
    int k = i / NC, n = i % NC;
    BT[n * K + k] = (_Float16)B[i];
}

// ---------------- WMMA GEMM: C[M x NC] = A[M x K] @ B[K x NC] ----------------
// A packed f16 row-major; B packed f16 transposed (NC x K). One 16x16 tile/wave.
// Fragment layouts (k = half*16 + s, same for A and B): A lane rows, B lane cols, K split at
// half-wave; both fragments are contiguous f16 runs -> pure B128 traffic.
__global__ void k_gemm_wmma(const _Float16* __restrict__ Ap, const _Float16* __restrict__ BT,
                            float* __restrict__ C, int M, int K, int NC,
                            int strideA, int strideC) {
    const int wave = threadIdx.x >> 5;
    const int lane = threadIdx.x & 31;
    const int tilesN = NC >> 4;
    const int tile = blockIdx.x * (blockDim.x >> 5) + wave;
    if (tile >= (M >> 4) * tilesN) return;   // wave-uniform: EXEC stays all-1s
    const int tm = tile / tilesN, tn = tile % tilesN;
    const _Float16* Ab = Ap + blockIdx.y * strideA;
    float*          Cb = C  + blockIdx.y * strideC;
    const int half = lane >> 4, r = lane & 15;
    const int row = (tm << 4) + r;
    const int col = (tn << 4) + r;
    v8f acc = {};
    for (int kb = 0; kb < K; kb += 32) {
        const _Float16* arow = Ab + row * K + kb + (half << 4);
        v8h alo = *(const v8h*)(arow);          // K = half*16 + 0..7
        v8h ahi = *(const v8h*)(arow + 8);      // K = half*16 + 8..15
        v16h a = __builtin_shufflevector(alo, ahi,
                                         0, 1, 2, 3, 4, 5, 6, 7,
                                         8, 9, 10, 11, 12, 13, 14, 15);
        v16h b = *(const v16h*)(BT + col * K + kb + (half << 4));  // K = half*16 + 0..15
        acc = __builtin_amdgcn_wmma_f32_16x16x32_f16(false, a, false, b,
                                                     (short)0, acc, false, false);
    }
#pragma unroll
    for (int v = 0; v < 8; ++v)   // C/D: lane = N, VGPR v -> M = half*8 + v
        Cb[((tm << 4) + (half << 3) + v) * NC + col] = acc[v];
}

// ---------------- GCN aggregation ----------------
__global__ void k_self(const float* __restrict__ xw, const float* __restrict__ invdeg,
                       float* __restrict__ agg, int F) {
    int i = blockIdx.x * 256 + threadIdx.x;
    if (i >= T_ * N_ * F) return;
    agg[i] = xw[i] * invdeg[i / F];        // i/F == t*N + n
}

__global__ void k_scatter(const float* __restrict__ xw, const int* __restrict__ ei,
                          const float* __restrict__ dinv, float* __restrict__ agg, int F) {
    const int chunks = F >> 2;
    int g = blockIdx.x * 256 + threadIdx.x;
    if (g >= T_ * E_ * chunks) return;
    const int c = g % chunks; g /= chunks;
    const int e = g % E_;     const int t = g / E_;
    const int* eit = ei + t * 2 * E_;
    const int src = eit[e], dst = eit[E_ + e];
    const float nrm = dinv[t * N_ + src] * dinv[t * N_ + dst];
    const float4 xv = *(const float4*)(xw + (t * N_ + src) * F + (c << 2));
    float* ad = agg + (t * N_ + dst) * F + (c << 2);
    atomicAdd(ad + 0, xv.x * nrm);
    atomicAdd(ad + 1, xv.y * nrm);
    atomicAdd(ad + 2, xv.z * nrm);
    atomicAdd(ad + 3, xv.w * nrm);
}

// layer1: h = relu(agg + b), emitted directly as packed f16 for the next WMMA GEMM
__global__ void k_bias_relu_pack(const float* __restrict__ a, const float* __restrict__ b,
                                 _Float16* __restrict__ hpack) {
    int i = blockIdx.x * 256 + threadIdx.x;
    if (i >= T_ * N_ * HID_) return;
    hpack[i] = (_Float16)fmaxf(a[i] + b[i % HID_], 0.0f);
}

// layer2: enc = agg + b (in place, f32)
__global__ void k_bias(float* __restrict__ a, const float* __restrict__ b, int F) {
    int i = blockIdx.x * 256 + threadIdx.x;
    if (i >= T_ * N_ * F) return;
    a[i] += b[i % F];
}

// ---------------- pooled mean over nodes ----------------
__global__ void k_pool(const float* __restrict__ enc, float* __restrict__ pooled) {
    __shared__ float red[256];
    const int t = blockIdx.x / OUT_, f = blockIdx.x % OUT_;
    float s = 0.0f;
    for (int n = threadIdx.x; n < N_; n += 256) s += enc[(t * N_ + n) * OUT_ + f];
    red[threadIdx.x] = s;
    __syncthreads();
    for (int w = 128; w > 0; w >>= 1) {
        if (threadIdx.x < w) red[threadIdx.x] += red[threadIdx.x + w];
        __syncthreads();
    }
    if (threadIdx.x == 0) pooled[t * OUT_ + f] = red[0] * (1.0f / N_);
}

// ---------------- GRU (single block, sequential over T) ----------------
__global__ void k_gru(const float* __restrict__ pooled, const float* __restrict__ Wih,
                      const float* __restrict__ Whh, const float* __restrict__ bih,
                      const float* __restrict__ bhh, float* __restrict__ hout) {
    __shared__ float h[HID_];
    __shared__ float gi[3 * HID_], gh[3 * HID_];
    const int tid = threadIdx.x;               // 0..383
    if (tid < HID_) h[tid] = 0.0f;
    __syncthreads();
    for (int t = 0; t < T_; ++t) {
        const float* xt = pooled + t * OUT_;
        float si = bih[tid], sh = bhh[tid];
        for (int k = 0; k < OUT_; ++k) si += Wih[tid * OUT_ + k] * xt[k];
        for (int k = 0; k < HID_; ++k) sh += Whh[tid * HID_ + k] * h[k];
        gi[tid] = si; gh[tid] = sh;
        __syncthreads();
        float hn = 0.0f;
        if (tid < HID_) {
            float rr = 1.0f / (1.0f + expf(-(gi[tid] + gh[tid])));
            float zz = 1.0f / (1.0f + expf(-(gi[HID_ + tid] + gh[HID_ + tid])));
            float nn = tanhf(gi[2 * HID_ + tid] + rr * gh[2 * HID_ + tid]);
            hn = (1.0f - zz) * nn + zz * h[tid];
        }
        __syncthreads();
        if (tid < HID_) h[tid] = hn;
        __syncthreads();
    }
    if (tid < HID_) hout[tid] = h[tid];
}

// ---------------- decoder c-vector and fold ----------------
__global__ void k_cvec(const float* __restrict__ hg, const float* __restrict__ Wd1,
                       const float* __restrict__ bd1, float* __restrict__ c) {
    int j = threadIdx.x;
    if (j >= HID_) return;
    float s = bd1[j];
    for (int k = 0; k < HID_; ++k) s += hg[k] * Wd1[(2 * OUT_ + k) * HID_ + j];
    c[j] = s;
}

__global__ void k_addc(float* __restrict__ A, const float* __restrict__ c) {
    int i = blockIdx.x * 256 + threadIdx.x;
    if (i >= N_ * HID_) return;
    A[i] += c[i % HID_];
}

// ---------------- TDM: load a 16 x HID f32 tile into LDS with row padding -----
#if __has_builtin(__builtin_amdgcn_tensor_load_to_lds)
typedef __attribute__((ext_vector_type(4))) unsigned int v4u;
typedef __attribute__((ext_vector_type(8))) int          v8i;
typedef __attribute__((ext_vector_type(4))) int          v4i;

__device__ __forceinline__ void tdm_load_tile16xHID(unsigned lds_off, const float* g) {
    unsigned long long ga = (unsigned long long)(size_t)g;
    // D# group0: count=1, lds_addr, 57-bit global addr, type=2 ("image")
    v4u g0 = { 1u, lds_off, (unsigned)ga,
               (unsigned)((ga >> 32) & 0x01FFFFFFu) | 0x80000000u };
    // D# group1: data_size=2 (4B), pad_enable=1, pad_interval=6 (128 DWORDs),
    // pad_amount=0 (1 DWORD) -> LDS row pitch = HID_+1 floats (bank-conflict free)
    const unsigned td0 = HID_, td1 = 16, tw = HID_, th = 16, st0 = HID_;
    v8i g1 = { (int)((2u << 16) | (1u << 20) | (6u << 22)),
               (int)((td0 & 0xFFFFu) << 16),
               (int)((td0 >> 16) | ((td1 & 0xFFFFu) << 16)),
               (int)((td1 >> 16) | (tw << 16)),
               (int)th,
               (int)st0,
               0, 0 };
    v4i z4 = { 0, 0, 0, 0 };
#if defined(__clang_major__) && (__clang_major__ >= 23)
    v8i z8 = { 0, 0, 0, 0, 0, 0, 0, 0 };
    __builtin_amdgcn_tensor_load_to_lds(g0, g1, z4, z4, z8, 0);
#else
    __builtin_amdgcn_tensor_load_to_lds(g0, g1, z4, z4, 0);
#endif
}
#define TDM_OK 1
#else
#define TDM_OK 0
#endif

// ---------------- all-pairs scorer: relu(A_i + B_j) . Wd2, off-diag only ------
__global__ void k_pairs(const float* __restrict__ Ab, const float* __restrict__ Bb,
                        const float* __restrict__ w2, const float* __restrict__ b2,
                        float* __restrict__ out) {
    __shared__ float sA[16][HID_ + 1];
    __shared__ float sB[16][HID_ + 1];
    __shared__ float sW[HID_];
    const int tid = threadIdx.x;
    const int i0 = blockIdx.y << 4, j0 = blockIdx.x << 4;
#if TDM_OK && __has_builtin(__builtin_amdgcn_s_wait_tensorcnt)
    if (tid < 32) {   // wave 0 issues both tile DMAs, waits, then WG barrier
        tdm_load_tile16xHID((unsigned)(size_t)(void*)&sA[0][0], Ab + i0 * HID_);
        tdm_load_tile16xHID((unsigned)(size_t)(void*)&sB[0][0], Bb + j0 * HID_);
        __builtin_amdgcn_s_wait_tensorcnt(0);
    }
#else
    for (int idx = tid; idx < 16 * HID_; idx += 256) {
        const int rr = idx >> 7, ff = idx & (HID_ - 1);
        sA[rr][ff] = Ab[(i0 + rr) * HID_ + ff];
        sB[rr][ff] = Bb[(j0 + rr) * HID_ + ff];
    }
#endif
    if (tid < HID_) sW[tid] = w2[tid];
    __syncthreads();
    const int ti = tid >> 4, tj = tid & 15;
    float s = 0.0f;
#pragma unroll 8
    for (int f = 0; f < HID_; ++f)
        s += fmaxf(sA[ti][f] + sB[tj][f], 0.0f) * sW[f];
    s += b2[0];
    const int i = i0 + ti, j = j0 + tj;
    if (i != j)
        out[i * (N_ - 1) + j - (j > i ? 1 : 0)] = s;
}

// ---------------- host launch ----------------
extern "C" void kernel_launch(void* const* d_in, const int* in_sizes, int n_in,
                              void* d_out, int out_size, void* d_ws, size_t ws_size,
                              hipStream_t stream) {
    (void)in_sizes; (void)n_in; (void)out_size; (void)ws_size;
    const float* x_seq = (const float*)d_in[0];
    const int*   ei    = (const int*)  d_in[1];
    const float* Wc1   = (const float*)d_in[2];
    const float* bc1   = (const float*)d_in[3];
    const float* Wc2   = (const float*)d_in[4];
    const float* bc2   = (const float*)d_in[5];
    const float* Wih   = (const float*)d_in[6];
    const float* Whh   = (const float*)d_in[7];
    const float* bih   = (const float*)d_in[8];
    const float* bhh   = (const float*)d_in[9];
    const float* Wd1   = (const float*)d_in[10];
    const float* bd1   = (const float*)d_in[11];
    const float* Wd2   = (const float*)d_in[12];
    const float* bd2   = (const float*)d_in[13];
    float* out = (float*)d_out;

    float* w = (float*)d_ws;
    float* deg    = w; w += T_ * N_;          // becomes 1/deg
    float* dinv   = w; w += T_ * N_;
    float* xw1    = w; w += T_ * N_ * HID_;
    float* agg1   = w; w += T_ * N_ * HID_;
    float* xw2    = w; w += T_ * N_ * OUT_;
    float* enc    = w; w += T_ * N_ * OUT_;
    float* pooled = w; w += T_ * OUT_;
    float* hgru   = w; w += HID_;
    float* Abuf   = w; w += N_ * HID_;
    float* Bbuf   = w; w += N_ * HID_;
    float* cvec   = w; w += HID_;
    _Float16* hp = (_Float16*)w;              // 32B-aligned (offset is mult. of 32B)
    _Float16* xpack = hp; hp += T_ * N_ * IN_;
    _Float16* hpack = hp; hp += T_ * N_ * HID_;
    _Float16* epack = hp; hp += N_ * OUT_;
    _Float16* Wc1T  = hp; hp += IN_ * HID_;
    _Float16* Wc2T  = hp; hp += HID_ * OUT_;
    _Float16* WaT   = hp; hp += OUT_ * HID_;
    _Float16* WbT   = hp; hp += OUT_ * HID_;

    auto nb = [](int n) { return (n + 255) / 256; };

    // degrees
    k_init_deg <<<nb(T_ * N_), 256, 0, stream>>>(deg);
    k_count_deg<<<nb(T_ * E_), 256, 0, stream>>>(ei, deg);
    k_fin_deg  <<<nb(T_ * N_), 256, 0, stream>>>(deg, dinv);

    // one-time f16 packing (weights transposed for contiguous B fragments)
    k_pack_f16<<<nb(T_ * N_ * IN_), 256, 0, stream>>>(x_seq, xpack, T_ * N_ * IN_);
    k_packBT  <<<nb(IN_ * HID_),  256, 0, stream>>>(Wc1, Wc1T, IN_, HID_);
    k_packBT  <<<nb(HID_ * OUT_), 256, 0, stream>>>(Wc2, Wc2T, HID_, OUT_);
    k_packBT  <<<nb(OUT_ * HID_), 256, 0, stream>>>(Wd1,              WaT, OUT_, HID_);
    k_packBT  <<<nb(OUT_ * HID_), 256, 0, stream>>>(Wd1 + OUT_ * HID_, WbT, OUT_, HID_);

    // GCN layer 1: xw1 = x @ Wc1 (1024x64x128, 512 tiles -> 128 blocks x 4 waves)
    k_gemm_wmma<<<dim3(128, T_), 128, 0, stream>>>(xpack, Wc1T, xw1, N_, IN_, HID_,
                                                   N_ * IN_, N_ * HID_);
    k_self   <<<nb(T_ * N_ * HID_), 256, 0, stream>>>(xw1, deg, agg1, HID_);
    k_scatter<<<nb(T_ * E_ * (HID_ / 4)), 256, 0, stream>>>(xw1, ei, dinv, agg1, HID_);
    k_bias_relu_pack<<<nb(T_ * N_ * HID_), 256, 0, stream>>>(agg1, bc1, hpack);

    // GCN layer 2: xw2 = h @ Wc2 (1024x128x64, 256 tiles -> 64 blocks)
    k_gemm_wmma<<<dim3(64, T_), 128, 0, stream>>>(hpack, Wc2T, xw2, N_, HID_, OUT_,
                                                  N_ * HID_, N_ * OUT_);
    k_self   <<<nb(T_ * N_ * OUT_), 256, 0, stream>>>(xw2, deg, enc, OUT_);
    k_scatter<<<nb(T_ * E_ * (OUT_ / 4)), 256, 0, stream>>>(xw2, ei, dinv, enc, OUT_);
    k_bias   <<<nb(T_ * N_ * OUT_), 256, 0, stream>>>(enc, bc2, OUT_);

    // pooling + GRU
    k_pool<<<T_ * OUT_, 256, 0, stream>>>(enc, pooled);
    k_gru <<<1, 3 * HID_, 0, stream>>>(pooled, Wih, Whh, bih, bhh, hgru);

    // decoder GEMMs on last embedding (1024x64x128 each)
    const float* last = enc + (T_ - 1) * N_ * OUT_;
    k_pack_f16<<<nb(N_ * OUT_), 256, 0, stream>>>(last, epack, N_ * OUT_);
    k_gemm_wmma<<<dim3(128, 1), 128, 0, stream>>>(epack, WaT, Abuf, N_, OUT_, HID_, 0, 0);
    k_gemm_wmma<<<dim3(128, 1), 128, 0, stream>>>(epack, WbT, Bbuf, N_, OUT_, HID_, 0, 0);
    k_cvec<<<1, HID_, 0, stream>>>(hgru, Wd1, bd1, cvec);
    k_addc<<<nb(N_ * HID_), 256, 0, stream>>>(Abuf, cvec);

    // all-pairs scoring (64x64 blocks of 16x16 pairs, TDM-staged LDS tiles)
    k_pairs<<<dim3(N_ / 16, N_ / 16), 256, 0, stream>>>(Abuf, Bbuf, Wd2, bd2, out);
}
